// CAM_27384711479300
// MI455X (gfx1250) — matrix operand order repaired and checked by reading, not compiled, not run on hardware
//
#include <hip/hip_runtime.h>

// ---------------------------------------------------------------------------
// CAM (channel attention) x=[16,64,64,512] fp32, gfx1250 (wave32, bf16 WMMA)
// Pipeline:
//   K0 pack:    xT[b][c][n] = bf16(x[b][n][c])          (LDS tile transpose)
//   K1 gram:    G = Xt * Xt^T   (upper-tri 64x64 tiles + mirror, 4x4 WMMA)
//   K2 softmax: PT[b][d][c] = softmax_d(G[b][c][:])     (transposed store)
//   K3 out:     out = gamma * (A @ P) + x               (4x4 WMMA tiles)
// All WMMA fragments are contiguous bf16 runs -> global_load_b128 pairs.
// ---------------------------------------------------------------------------

typedef __attribute__((ext_vector_type(16))) __bf16 v16bf;
typedef __attribute__((ext_vector_type(8)))  __bf16 v8bf;
typedef __attribute__((ext_vector_type(4)))  __bf16 v4bf;
typedef __attribute__((ext_vector_type(8)))  float  v8f;
typedef __attribute__((ext_vector_type(4)))  float  v4f;

static constexpr int B_  = 16;
static constexpr int C_  = 512;
static constexpr int HW_ = 4096;

__device__ __forceinline__ __bf16 f2bf(float f) {
    return static_cast<__bf16>(f);   // v_cvt_pk_bf16_f32 when paired
}

__device__ __forceinline__ v8f wmma_bf16(v16bf a, v16bf b, v8f c) {
    return __builtin_amdgcn_wmma_f32_16x16x32_bf16(false, a, false, b,
                                                   (short)0, c, false, false);
}

// ---------------------------------------------------------------------------
// K0: pack + transpose  xT[b][c][n] = bf16(x[b][n][c]).
// One block per 64(n) x 64(c) tile; coalesced b128 in, b64 out via LDS.
// ---------------------------------------------------------------------------
__global__ void __launch_bounds__(256)
cam_pack(const float* __restrict__ x, __bf16* __restrict__ xT) {
    __shared__ __bf16 tile[64][72];   // [c][n], padded stride vs bank conflicts

    const int blk  = (int)blockIdx.x;
    const int b    = blk >> 9;
    const int rest = blk & 511;
    const int n0   = (rest >> 3) << 6;
    const int c0   = (rest & 7)  << 6;
    const int tid  = (int)threadIdx.x;

    const float* __restrict__ xb = x + (size_t)b * HW_ * C_;

#pragma unroll
    for (int i = 0; i < 4; ++i) {                // 1024 float4 / 256 threads
        const int flat = tid + 256 * i;
        const int n = flat >> 4;                 // 0..63
        const int c = (flat & 15) << 2;          // 0..60 step 4
        const v4f v = *(const v4f*)(xb + (size_t)(n0 + n) * C_ + c0 + c);
#pragma unroll
        for (int j = 0; j < 4; ++j)
            tile[c + j][n] = f2bf(v[j]);
    }
    __syncthreads();

    __bf16* __restrict__ xTb = xT + (size_t)b * C_ * HW_;
#pragma unroll
    for (int i = 0; i < 4; ++i) {
        const int flat = tid + 256 * i;
        const int cc = flat >> 4;
        const int nn = (flat & 15) << 2;
        const v4bf v = *(const v4bf*)&tile[cc][nn];
        *(v4bf*)(xTb + (size_t)(c0 + cc) * HW_ + n0 + nn) = v;
    }
}

// ---------------------------------------------------------------------------
// K1: Gram  G[b,c,d] = sum_n xT[b,c,n]*xT[b,d,n]  (symmetric).
// One wave per upper-triangular 64x64 tile (36/batch), 4x4 WMMA grid,
// mirror-store the transposed tile. All fragments contiguous bf16.
// ---------------------------------------------------------------------------
__global__ void __launch_bounds__(256)
cam_gram(const __bf16* __restrict__ xT, float* __restrict__ G) {
    const int wave = (int)((blockIdx.x * blockDim.x + threadIdx.x) >> 5);
    const int lane = (int)(threadIdx.x & 31);
    const int b = wave / 36;
    int rem = wave % 36, ti = 0;
    while (rem >= (8 - ti)) { rem -= (8 - ti); ++ti; }
    const int tj = ti + rem;
    const int c0 = ti << 6;
    const int d0 = tj << 6;

    const __bf16* __restrict__ xb = xT + (size_t)b * C_ * HW_;

    const int lm  = lane & 15;
    const int kbA = (lane < 16) ? 0 : 16;
    const int kbB = (lane < 16) ? 0 : 16;

    v8f acc[4][4];
#pragma unroll
    for (int i = 0; i < 4; ++i)
#pragma unroll
        for (int j = 0; j < 4; ++j) acc[i][j] = (v8f){0,0,0,0,0,0,0,0};

    for (int k0 = 0; k0 < HW_; k0 += 32) {
        v16bf bf[4];
#pragma unroll
        for (int nt = 0; nt < 4; ++nt)           // B: row d, 32B contiguous
            bf[nt] = *(const v16bf*)(xb + (size_t)(d0 + nt * 16 + lm) * HW_ +
                                     k0 + kbB);
#pragma unroll
        for (int mt = 0; mt < 4; ++mt) {         // A: row c, 2x16B runs
            const __bf16* ar = xb + (size_t)(c0 + mt * 16 + lm) * HW_ + k0;
            const v8bf lo = *(const v8bf*)(ar + kbA);
            const v8bf hi = *(const v8bf*)(ar + 8 + kbA);
            v16bf af;
#pragma unroll
            for (int i = 0; i < 8; ++i) { af[i] = lo[i]; af[8 + i] = hi[i]; }
#pragma unroll
            for (int nt = 0; nt < 4; ++nt)
                acc[mt][nt] = wmma_bf16(af, bf[nt], acc[mt][nt]);
        }
    }

    float* __restrict__ Gb = G + (size_t)b * C_ * C_;
    const int mrow = (lane < 16) ? 0 : 8;
#pragma unroll
    for (int mt = 0; mt < 4; ++mt)
#pragma unroll
        for (int nt = 0; nt < 4; ++nt)
#pragma unroll
            for (int v = 0; v < 8; ++v) {
                const int r  = c0 + mt * 16 + mrow + v;
                const int cc = d0 + nt * 16 + lm;
                const float val = acc[mt][nt][v];
                Gb[(size_t)r * C_ + cc] = val;   // upper tile
                Gb[(size_t)cc * C_ + r] = val;   // mirrored tile
            }
}

// ---------------------------------------------------------------------------
// K2: row softmax over G (512 elems) -> transposed bf16 PT[b][d][c].
// One wave per row c; shfl_xor tree reductions (wave32).
// ---------------------------------------------------------------------------
__global__ void __launch_bounds__(256)
cam_softmax(const float* __restrict__ G, __bf16* __restrict__ PT) {
    const int wave = (int)((blockIdx.x * blockDim.x + threadIdx.x) >> 5);
    const int lane = (int)(threadIdx.x & 31);
    const int b = wave >> 9;
    const int c = wave & 511;
    const float* __restrict__ row = G + (size_t)wave * C_;

    float v[16];
    float m = -3.4e38f;
#pragma unroll
    for (int i = 0; i < 16; ++i) {
        v[i] = row[lane + 32 * i];
        m = fmaxf(m, v[i]);
    }
#pragma unroll
    for (int off = 16; off >= 1; off >>= 1)
        m = fmaxf(m, __shfl_xor(m, off, 32));

    float s = 0.0f;
#pragma unroll
    for (int i = 0; i < 16; ++i) {
        v[i] = __expf(v[i] - m);
        s += v[i];
    }
#pragma unroll
    for (int off = 16; off >= 1; off >>= 1)
        s += __shfl_xor(s, off, 32);

    const float inv = 1.0f / s;
    __bf16* __restrict__ PTb = PT + (size_t)b * C_ * C_;
#pragma unroll
    for (int i = 0; i < 16; ++i) {
        const int d = lane + 32 * i;
        PTb[(size_t)d * C_ + c] = f2bf(v[i] * inv);   // transposed store
    }
}

// ---------------------------------------------------------------------------
// K3: Out[b,n,d] = gamma * sum_c x[b,n,c]*P[b,c,d] + x[b,n,d]
// One wave per 64x64 tile (4x4 WMMA). A from x rows (v8f loads + cvt),
// B from PT rows (32B contiguous bf16).
// ---------------------------------------------------------------------------
__global__ void __launch_bounds__(256)
cam_out(const float* __restrict__ x, const __bf16* __restrict__ PT,
        const float* __restrict__ gamma, float* __restrict__ out) {
    const int wave = (int)((blockIdx.x * blockDim.x + threadIdx.x) >> 5);
    const int lane = (int)(threadIdx.x & 31);
    const int b  = wave >> 9;
    const int t  = wave & 511;
    const int m0 = (t >> 3) << 6;
    const int d0 = (t & 7)  << 6;

    const float*  __restrict__ xb  = x  + (size_t)b * HW_ * C_;
    const __bf16* __restrict__ PTb = PT + (size_t)b * C_ * C_;

    const int lm  = lane & 15;
    const int kbA = (lane < 16) ? 0 : 16;
    const int kbB = (lane < 16) ? 0 : 16;

    v8f acc[4][4];
#pragma unroll
    for (int i = 0; i < 4; ++i)
#pragma unroll
        for (int j = 0; j < 4; ++j) acc[i][j] = (v8f){0,0,0,0,0,0,0,0};

    for (int k0 = 0; k0 < C_; k0 += 32) {
        v16bf bf[4];
#pragma unroll
        for (int nt = 0; nt < 4; ++nt)           // B = PT row d, contiguous
            bf[nt] = *(const v16bf*)(PTb + (size_t)(d0 + nt * 16 + lm) * C_ +
                                     k0 + kbB);
#pragma unroll
        for (int mt = 0; mt < 4; ++mt) {         // A = x row n, contiguous f32
            const float* r = xb + (size_t)(m0 + mt * 16 + lm) * C_ + k0;
            const v8f flo = *(const v8f*)(r + kbA);
            const v8f fhi = *(const v8f*)(r + 8 + kbA);
            v16bf af;
#pragma unroll
            for (int i = 0; i < 8; ++i) {
                af[i]     = f2bf(flo[i]);
                af[8 + i] = f2bf(fhi[i]);
            }
#pragma unroll
            for (int nt = 0; nt < 4; ++nt)
                acc[mt][nt] = wmma_bf16(af, bf[nt], acc[mt][nt]);
        }
    }

    const float g = gamma[0];
    const int mrow = (lane < 16) ? 0 : 8;
    float* __restrict__ ob = out + (size_t)b * HW_ * C_;
#pragma unroll
    for (int mt = 0; mt < 4; ++mt)
#pragma unroll
        for (int nt = 0; nt < 4; ++nt)
#pragma unroll
            for (int v = 0; v < 8; ++v) {
                const size_t idx = (size_t)(m0 + mt * 16 + mrow + v) * C_ +
                                   (d0 + nt * 16 + lm);
                ob[idx] = g * acc[mt][nt][v] + xb[idx];
            }
}

// ---------------------------------------------------------------------------
extern "C" void kernel_launch(void* const* d_in, const int* in_sizes, int n_in,
                              void* d_out, int out_size, void* d_ws, size_t ws_size,
                              hipStream_t stream) {
    const float* x     = (const float*)d_in[0];
    const float* gamma = (const float*)d_in[1];
    float*       out   = (float*)d_out;

    // Workspace: G fp32 (16 MiB) | PT bf16 (8 MiB) | xT bf16 (64 MiB)
    float*  G  = (float*)d_ws;
    __bf16* PT = (__bf16*)((char*)d_ws + (size_t)16 * 1024 * 1024);
    __bf16* xT = (__bf16*)((char*)d_ws + (size_t)24 * 1024 * 1024);

    // K0: 16 batches * 512 tiles
    cam_pack<<<B_ * 512, 256, 0, stream>>>(x, xT);

    // K1: 16 * 36 upper-tri tiles = 576 waves
    cam_gram<<<(B_ * 36 * 32) / 256, 256, 0, stream>>>(xT, G);

    // K2: 8192 rows, one wave each
    cam_softmax<<<(B_ * C_ * 32) / 256, 256, 0, stream>>>(G, PT);

    // K3: 16 * 512 tiles = 8192 waves
    cam_out<<<(B_ * 512 * 32) / 256, 256, 0, stream>>>(x, PT, gamma, out);
}
